// MCSoftContrastiveLoss_36189394436352
// MI455X (gfx1250) — matrix operand order repaired and checked by reading, not compiled, not run on hardware
//
#include <hip/hip_runtime.h>
#include <math.h>

typedef __attribute__((ext_vector_type(16))) _Float16 v16h;
typedef __attribute__((ext_vector_type(8)))  _Float16 v8h;
typedef __attribute__((ext_vector_type(4)))  _Float16 v4h;
typedef __attribute__((ext_vector_type(8)))  float    v8f;
typedef __attribute__((ext_vector_type(4)))  float    v4f;

#define NROW 4096        // N*K flattened rows
#define DDIM 1024        // reduction dim
#define KC   64          // K-chunk staged in LDS (elements) -- double buffered
#define LDSS 72          // LDS row stride in halves: KC + 8 pad -> 144B (+36 banks/row, conflict-free)

__device__ __forceinline__ v16h cat8(v8h lo, v8h hi) {
  return __builtin_shufflevector(lo, hi, 0,1,2,3,4,5,6,7,8,9,10,11,12,13,14,15);
}

// -------- per-row squared norms of x (rows 0..4095) and y (rows 4096..8191) --------
__global__ __launch_bounds__(256) void mcsoft_norms(const float* __restrict__ x,
                                                    const float* __restrict__ y,
                                                    float* __restrict__ sq) {
  int gwave = (blockIdx.x * 256 + threadIdx.x) >> 5;
  int lane  = threadIdx.x & 31;
  if (gwave >= 2 * NROW) return;
  const float* src = (gwave < NROW) ? (x + (size_t)gwave * DDIM)
                                    : (y + (size_t)(gwave - NROW) * DDIM);
  const v4f* s4 = (const v4f*)src;
  float acc = 0.f;
#pragma unroll
  for (int t = 0; t < DDIM / 4 / 32; ++t) {
    v4f v = s4[lane + 32 * t];
    acc += v.x * v.x + v.y * v.y + v.z * v.z + v.w * v.w;
  }
#pragma unroll
  for (int m = 16; m >= 1; m >>= 1) acc += __shfl_xor(acc, m, 32);
  if (lane == 0) sq[gwave] = acc;
}

// -------- fused GEMM (WMMA f16 -> f32) + soft-contrastive NLL epilogue --------
// Block = 64x64 macro-tile = 4x4 (i,j) pairs; 8 waves, 2 WMMA tiles per wave.
// Double-buffered LDS staging; per-block deterministic partial sum to d_ws.
__global__ __launch_bounds__(256) void mcsoft_loss(const float* __restrict__ x,
                                                   const float* __restrict__ y,
                                                   const float* __restrict__ sq,
                                                   const float* __restrict__ nsp,
                                                   const float* __restrict__ shp,
                                                   float* __restrict__ part) {
  __shared__ __align__(16) _Float16 ldsA[2][64][LDSS];
  __shared__ __align__(16) _Float16 ldsB[2][64][LDSS];
  __shared__ float wsum[8];

  const int tid    = threadIdx.x;
  const int lane   = tid & 31;
  const int wave   = tid >> 5;
  const int ti     = wave >> 1;        // 0..3 : i-tile within block
  const int tjb    = (wave & 1) * 2;   // 0 or 2: base j-tile (wave owns 2)
  const int laneHi = lane >> 4;
  const int mn     = lane & 15;

  const int rowA0 = blockIdx.x * 64;   // 4 i-values * 16 rows
  const int rowB0 = blockIdx.y * 64;

  const v4f* x4 = (const v4f*)x;
  const v4f* y4 = (const v4f*)y;

  v8f c0 = {}; v8f c1 = {};
  v4f ra[4], rb[4];

  // prologue: prefetch chunk 0 into registers
#pragma unroll
  for (int p = 0; p < 4; ++p) {
    int idx = p * 256 + tid;           // 0..1023
    int row = idx >> 4;                // 0..63
    int c4  = idx & 15;                // float4 column within 64-elem chunk
    ra[p] = x4[(size_t)(rowA0 + row) * (DDIM / 4) + c4];
    rb[p] = y4[(size_t)(rowB0 + row) * (DDIM / 4) + c4];
  }

  for (int kc = 0; kc < DDIM / KC; ++kc) {
    const int buf = kc & 1;
    // ---- convert prefetched chunk to f16 and store to LDS ----
#pragma unroll
    for (int p = 0; p < 4; ++p) {
      int idx = p * 256 + tid;
      int row = idx >> 4;
      int c4  = idx & 15;
      v4h ha = { (_Float16)ra[p].x, (_Float16)ra[p].y, (_Float16)ra[p].z, (_Float16)ra[p].w };
      v4h hb = { (_Float16)rb[p].x, (_Float16)rb[p].y, (_Float16)rb[p].z, (_Float16)rb[p].w };
      *(v4h*)&ldsA[buf][row][c4 * 4] = ha;
      *(v4h*)&ldsB[buf][row][c4 * 4] = hb;
    }
    __syncthreads();   // single barrier per iter: double buffer covers WAR across iters

    // ---- issue next chunk's global loads; WMMA below hides their latency ----
    if (kc + 1 < DDIM / KC) {
#pragma unroll
      for (int p = 0; p < 4; ++p) {
        int idx = p * 256 + tid;
        int row = idx >> 4;
        int c4  = idx & 15;
        ra[p] = x4[(size_t)(rowA0 + row) * (DDIM / 4) + (kc + 1) * (KC / 4) + c4];
        rb[p] = y4[(size_t)(rowB0 + row) * (DDIM / 4) + (kc + 1) * (KC / 4) + c4];
      }
    }

    // ---- 2 WMMA k-steps of 32 over this chunk ----
#pragma unroll
    for (int ks = 0; ks < KC / 32; ++ks) {
      const int base = ks * 32;
      // A 16x32 f16 frag: row M=mn, contiguous K (0-15 | 16-31 by lane half)
      v8h alo = *(const v8h*)&ldsA[buf][ti * 16 + mn][base + laneHi * 16];
      v8h ahi = *(const v8h*)&ldsA[buf][ti * 16 + mn][base + laneHi * 16 + 8];
      v16h a  = cat8(alo, ahi);
      // B 32x16 f16 frag: column N=mn, contiguous K (0-15 | 16-31 by lane half)
      v8h b0lo = *(const v8h*)&ldsB[buf][tjb * 16 + mn][base + laneHi * 16];
      v8h b0hi = *(const v8h*)&ldsB[buf][tjb * 16 + mn][base + laneHi * 16 + 8];
      v8h b1lo = *(const v8h*)&ldsB[buf][(tjb + 1) * 16 + mn][base + laneHi * 16];
      v8h b1hi = *(const v8h*)&ldsB[buf][(tjb + 1) * 16 + mn][base + laneHi * 16 + 8];
      v16h b0 = cat8(b0lo, b0hi);
      v16h b1 = cat8(b1lo, b1hi);
      c0 = __builtin_amdgcn_wmma_f32_16x16x32_f16(false, a, false, b0, (short)0, c0, false, false);
      c1 = __builtin_amdgcn_wmma_f32_16x16x32_f16(false, a, false, b1, (short)0, c1, false, false);
    }
  }

  // ---- epilogue: dist -> logits -> inner -> per-(i,j) logsumexp -> nll ----
  const float ns = nsp[0];
  const float sh = shp[0];
  const float* x2 = sq;
  const float* y2 = sq + NROW;
  const int i = blockIdx.x * 4 + ti;

  float xr[8];
#pragma unroll
  for (int r = 0; r < 8; ++r) xr[r] = x2[i * 16 + r + laneHi * 8];

  float nllsum = 0.f;
#pragma unroll
  for (int t = 0; t < 2; ++t) {
    v8f c = t ? c1 : c0;
    const int j = blockIdx.y * 4 + tjb + t;
    const float m = (i == j) ? 1.f : -1.f;
    const float yc = y2[j * 16 + mn];     // column N = mn of this tile
    float inner[8];
    float mx = -1e30f;
#pragma unroll
    for (int r = 0; r < 8; ++r) {
      float s2 = xr[r] + yc - 2.0f * c[r];
      s2 = fmaxf(s2, 0.0f);
      float d  = sqrtf(s2 + 1e-6f);
      float l  = sh - ns * d;             // logit
      float al = fabsf(l);
      float v  = m * l - al - log1pf(__expf(-2.0f * al));  // l*m - logaddexp(l,-l)
      inner[r] = v;
      mx = fmaxf(mx, v);
    }
#pragma unroll
    for (int s = 16; s >= 1; s >>= 1) mx = fmaxf(mx, __shfl_xor(mx, s, 32));
    float ss = 0.f;
#pragma unroll
    for (int r = 0; r < 8; ++r) ss += __expf(inner[r] - mx);
#pragma unroll
    for (int s = 16; s >= 1; s >>= 1) ss += __shfl_xor(ss, s, 32);
    float lse = mx + __logf(ss);
    nllsum += (-lse + 5.545177444479562f);   // + log(256)
  }

  // ---- deterministic block reduction: fixed wave order ----
  if (lane == 0) wsum[wave] = nllsum;
  __syncthreads();
  if (tid == 0) {
    float s = 0.f;
#pragma unroll
    for (int w = 0; w < 8; ++w) s += wsum[w];
    part[blockIdx.y * 64 + blockIdx.x] = s;
  }
}

// -------- deterministic final reduction of 4096 block partials --------
__global__ __launch_bounds__(256) void mcsoft_reduce(const float* __restrict__ part,
                                                     float* __restrict__ out) {
  __shared__ float s[256];
  const int t = threadIdx.x;
  float a = 0.f;
  for (int q = t; q < 4096; q += 256) a += part[q];   // fixed strided order
  s[t] = a;
  __syncthreads();
  for (int w = 128; w >= 1; w >>= 1) {                // fixed tree
    if (t < w) s[t] += s[t + w];
    __syncthreads();
  }
  if (t == 0) out[0] = 2.0f * s[0];                   // i2t sum == t2i sum
}

extern "C" void kernel_launch(void* const* d_in, const int* in_sizes, int n_in,
                              void* d_out, int out_size, void* d_ws, size_t ws_size,
                              hipStream_t stream) {
  const float* x  = (const float*)d_in[0];  // image_features   (256,16,1024)
  const float* y  = (const float*)d_in[1];  // caption_features (256,16,1024)
  // d_in[2], d_in[3] = logsigmas: unused (uniform_lambda = vib_beta = 0)
  const float* ns = (const float*)d_in[4];  // negative_scale (1,)
  const float* sh = (const float*)d_in[5];  // shift (1,)
  float* sq   = (float*)d_ws;               // 8192 floats: row sq-norms
  float* part = sq + 2 * NROW;              // 4096 floats: block partials (48KB total ws)
  float* out  = (float*)d_out;

  mcsoft_norms<<<dim3(1024), dim3(256), 0, stream>>>(x, y, sq);
  mcsoft_loss<<<dim3(64, 64), dim3(256), 0, stream>>>(x, y, sq, ns, sh, part);
  mcsoft_reduce<<<dim3(1), dim3(256), 0, stream>>>(part, out);
}
